// PillarAttention_73967926771831
// MI455X (gfx1250) — matrix operand order repaired and checked by reading, not compiled, not run on hardware
//
#include <hip/hip_runtime.h>
#include <math.h>

// ---------------- problem constants (fixed by the reference) ----------------
#define NTOT  8192
#define BATCH 4
#define SMAX  2816
#define CINC  64
#define EDIM  128
#define NHEAD 8
#define HDIM  16
#define E3    384
#define FF    256
#define QTMAX 176          // ceil(SMAX/16) 16-query tiles per batch
#define QG2   11           // (QTMAX/2 q-tile-pairs) / 8 waves-per-block

typedef __bf16 bf16;
typedef bf16  v16bf __attribute__((ext_vector_type(16)));
typedef bf16  v8bf  __attribute__((ext_vector_type(8)));
typedef float v8f   __attribute__((ext_vector_type(8)));

__device__ __forceinline__ v8f wmma_bf16(v16bf a, v16bf b, v8f c) {
  return __builtin_amdgcn_wmma_f32_16x16x32_bf16(false, a, false, b, (short)0, c,
                                                 false, false);
}

__device__ __forceinline__ v16bf frag_cat(v8bf lo, v8bf hi) {
  v16bf r;
#pragma unroll
  for (int e = 0; e < 8; ++e) { r[e] = lo[e]; r[e + 8] = hi[e]; }
  return r;
}

__device__ __forceinline__ float gelu_f(float v) {
  return 0.5f * v * (1.0f + erff(v * 0.70710678118654752f));
}

// ---------------- weight/feature bf16 conversion (+ W^T) --------------------
__global__ void k_prep(const float* __restrict__ pf,   const float* __restrict__ wpre,
                       const float* __restrict__ win,  const float* __restrict__ wout,
                       const float* __restrict__ w1,   const float* __restrict__ w2,
                       bf16* pf_b, bf16* wpre_t, bf16* win_t, bf16* wout_t,
                       bf16* w1_t, bf16* w2_t) {
  int t0 = blockIdx.x * blockDim.x + threadIdx.x;
  int stride = gridDim.x * blockDim.x;
  for (int t = t0; t < NTOT * CINC; t += stride) pf_b[t] = (bf16)pf[t];
  for (int t = t0; t < CINC * EDIM; t += stride) {           // [K][N] -> [N][K]
    int k = t / EDIM, n = t % EDIM; wpre_t[n * CINC + k] = (bf16)wpre[t];
  }
  for (int t = t0; t < EDIM * E3; t += stride) {
    int k = t / E3, n = t % E3;     win_t[n * EDIM + k] = (bf16)win[t];
  }
  for (int t = t0; t < EDIM * EDIM; t += stride) {
    int k = t / EDIM, n = t % EDIM; wout_t[n * EDIM + k] = (bf16)wout[t];
  }
  for (int t = t0; t < EDIM * FF; t += stride) {
    int k = t / FF, n = t % FF;     w1_t[n * EDIM + k] = (bf16)w1[t];
  }
  for (int t = t0; t < FF * EDIM; t += stride) {
    int k = t / EDIM, n = t % EDIM; w2_t[n * FF + k] = (bf16)w2[t];
  }
}

// ---------------- per-batch segment bounds ----------------------------------
// seg[0..3] = counts, seg[4..7] = starts
__global__ void k_seg(const int* __restrict__ vc, int* __restrict__ seg) {
  __shared__ int c[BATCH];
  if (threadIdx.x < BATCH) c[threadIdx.x] = 0;
  __syncthreads();
  for (int i = threadIdx.x; i < NTOT; i += blockDim.x)
    atomicAdd(&c[vc[i * 4]], 1);
  __syncthreads();
  if (threadIdx.x == 0) {
    int s = 0;
    for (int b = 0; b < BATCH; ++b) { seg[b] = c[b]; seg[4 + b] = s; s += c[b]; }
  }
}

// ---------------- generic bf16 WMMA GEMM: out = act(A @ Wt^T + bias) --------
// A: [M][K] bf16 row-major.  Wt: [Nn][K] bf16 (pre-transposed weights).
// One wave computes a 32(M) x 64(N) tile: per k-step, 2 A frags + 4 B frags
// are ALL loaded before the 8 WMMAs so the loads clause together and the
// waits stagger instead of fully draining before each matrix op.
__global__ void k_gemm(const bf16* __restrict__ A, const bf16* __restrict__ Wt,
                       const float* __restrict__ bias,
                       float* __restrict__ outf, bf16* __restrict__ outb,
                       int M, int Nn, int K, int act) {
  int gw  = (blockIdx.x * blockDim.x + threadIdx.x) >> 5;
  int ntn = Nn >> 6;                               // 64-wide N supertiles
  int mt  = gw / ntn, nt4 = gw % ntn;
  if (mt * 32 >= M) return;
  int lane = threadIdx.x & 31, ln = lane & 15, half = lane >> 4;

  const bf16* Ap0 = A  + (size_t)(mt * 32 + ln) * K + half * 8;
  const bf16* Ap1 = Ap0 + (size_t)16 * K;
  const bf16* Bp0 = Wt + (size_t)(nt4 * 64 + ln) * K + half * 8;
  v8f acc[2][4];
#pragma unroll
  for (int m = 0; m < 2; ++m)
#pragma unroll
    for (int t = 0; t < 4; ++t)
#pragma unroll
      for (int r = 0; r < 8; ++r) acc[m][t][r] = 0.0f;

  for (int k = 0; k < K; k += 32) {
    // ---- issue ALL loads for this k-step up front (distinct registers) ----
    v8bf a0lo = *(const v8bf*)(Ap0 + k);
    v8bf a0hi = *(const v8bf*)(Ap0 + k + 16);
    v8bf a1lo = *(const v8bf*)(Ap1 + k);
    v8bf a1hi = *(const v8bf*)(Ap1 + k + 16);
    v8bf b0[4], b1[4];
#pragma unroll
    for (int t = 0; t < 4; ++t) {
      const bf16* Bp = Bp0 + (size_t)t * 16 * K;
      b0[t] = *(const v8bf*)(Bp + k);
      b1[t] = *(const v8bf*)(Bp + k + 16);
    }
    // ---- 8 WMMAs reusing the two A fragments --------------------------------
    v16bf av0 = frag_cat(a0lo, a0hi);
    v16bf av1 = frag_cat(a1lo, a1hi);
#pragma unroll
    for (int t = 0; t < 4; ++t) {
      v16bf bv = frag_cat(b0[t], b1[t]);
      acc[0][t] = wmma_bf16(av0, bv, acc[0][t]);
      acc[1][t] = wmma_bf16(av1, bv, acc[1][t]);
    }
  }

#pragma unroll
  for (int m = 0; m < 2; ++m)
#pragma unroll
    for (int t = 0; t < 4; ++t) {
      int   col = nt4 * 64 + t * 16 + ln;
      float bv_ = bias[col];
#pragma unroll
      for (int r = 0; r < 8; ++r) {
        float v = acc[m][t][r] + bv_;
        if (act == 1) v = gelu_f(v);
        int    grow = mt * 32 + m * 16 + r + half * 8;   // C layout row
        size_t off  = (size_t)grow * Nn + col;
        if (outf) outf[off] = v;
        if (outb) outb[off] = (bf16)v;
      }
    }
}

// ---------------- V transpose: vt[c][b*S + local] = qkv[i][256+c] -----------
__global__ void k_vt(const bf16* __restrict__ qkv, const int* __restrict__ vc,
                     const int* __restrict__ seg, bf16* __restrict__ vt) {
  int t = blockIdx.x * blockDim.x + threadIdx.x;
  if (t >= NTOT * EDIM) return;
  int i = t / EDIM, c = t % EDIM;            // c == h*16 + d
  int b = vc[i * 4];
  int local = i - seg[4 + b];
  vt[(size_t)c * (BATCH * SMAX) + b * SMAX + local] =
      qkv[(size_t)i * E3 + 256 + c];
}

// -------- flash attention: one wave per (batch, head, pair of 16-q tiles) ---
// K/V fragments are loaded once per key tile and reused by both q-tiles.
__global__ void k_attn(const bf16* __restrict__ qkv, const bf16* __restrict__ vt,
                       const int* __restrict__ seg, bf16* __restrict__ ctx) {
  __shared__ __align__(16) bf16 pls[8][16][16];   // per-wave P tile staging
  int wave = threadIdx.x >> 5, lane = threadIdx.x & 31;
  int ln = lane & 15, half = lane >> 4;
  int b  = blockIdx.x / (NHEAD * QG2);
  int rm = blockIdx.x % (NHEAD * QG2);
  int h  = rm / QG2, qg = rm % QG2;
  int qt0 = (qg * 8 + wave) * 2;                  // first of two q-tiles
  int cnt = seg[b], st = seg[4 + b];
  if (qt0 * 16 >= cnt) return;

  // Q fragments (pre-scaled by 1/sqrt(HD)=0.25, exact in bf16)
  v16bf qa[2]; v8f o[2]; float mr[2][8], lr[2][8];
#pragma unroll
  for (int qi = 0; qi < 2; ++qi) {
    int qr = (qt0 + qi) * 16 + ln; if (qr >= cnt) qr = cnt - 1;
    const bf16* qp = qkv + (size_t)(st + qr) * E3 + h * 16 + half * 8;
    v8bf q8 = *(const v8bf*)qp;
#pragma unroll
    for (int e = 0; e < 8; ++e) {
      qa[qi][e]     = (bf16)((float)q8[e] * 0.25f);
      qa[qi][e + 8] = (bf16)0.0f;                 // K-dim 16..31 padding
    }
#pragma unroll
    for (int r = 0; r < 8; ++r) { o[qi][r] = 0.0f; mr[qi][r] = -1e30f; lr[qi][r] = 0.0f; }
  }

  int nkt = (cnt + 15) >> 4;
  for (int kt = 0; kt < nkt; ++kt) {
    int k0 = kt << 4;
    // K^T B-fragment: half h loads hd 8h..8h+7 of key (k0+ln) into e<8; e>=8 = pad
    v16bf kb;
#pragma unroll
    for (int e = 0; e < 16; ++e) kb[e] = (bf16)0.0f;
    // V B-fragment: half h loads keys 8h..8h+7 of hd column ln into e<8; e>=8 = pad
    v16bf vb;
#pragma unroll
    for (int e = 0; e < 16; ++e) vb[e] = (bf16)0.0f;
    {
      int key = k0 + ln; if (key >= cnt) key = cnt - 1;
      const bf16* kp = qkv + (size_t)(st + key) * E3 + 128 + h * 16 + half * 8;
      v8bf k0v = *(const v8bf*)kp;
      const bf16* vp = vt + (size_t)(h * 16 + ln) * (BATCH * SMAX) + b * SMAX + k0 + half * 8;
      v8bf v0 = *(const v8bf*)vp;
#pragma unroll
      for (int e = 0; e < 8; ++e) {
        kb[e] = k0v[e];
        vb[e] = v0[e];
      }
    }
    if (half == 0) {
      if (kt + 1 < nkt) {                                  // prefetch next tile
        int nk = k0 + 16 + ln; if (nk >= cnt) nk = cnt - 1;
        __builtin_prefetch(qkv + (size_t)(st + nk) * E3 + 128 + h * 16, 0, 0);
      }
    }

    bool kvalid = (k0 + ln) < cnt;          // col index = lane&15 in both halves
#pragma unroll
    for (int qi = 0; qi < 2; ++qi) {
      v8f s;
#pragma unroll
      for (int r = 0; r < 8; ++r) s[r] = 0.0f;
      s = wmma_bf16(qa[qi], kb, s);

#pragma unroll
      for (int r = 0; r < 8; ++r) {
        float sv = kvalid ? s[r] : -1e30f;
        float mx = sv;                      // row max within 16-lane half
        mx = fmaxf(mx, __shfl_xor(mx, 1, 32));
        mx = fmaxf(mx, __shfl_xor(mx, 2, 32));
        mx = fmaxf(mx, __shfl_xor(mx, 4, 32));
        mx = fmaxf(mx, __shfl_xor(mx, 8, 32));
        float mnew = fmaxf(mr[qi][r], mx);
        float corr = __expf(mr[qi][r] - mnew);
        float p    = __expf(sv - mnew);
        float rs = p;
        rs += __shfl_xor(rs, 1, 32);
        rs += __shfl_xor(rs, 2, 32);
        rs += __shfl_xor(rs, 4, 32);
        rs += __shfl_xor(rs, 8, 32);
        lr[qi][r] = lr[qi][r] * corr + rs;
        o[qi][r] *= corr;
        mr[qi][r] = mnew;
        pls[wave][r + half * 8][ln] = (bf16)p;   // C layout -> LDS row-major
      }

      // P A-fragment from LDS (same-wave DS ordering; no barrier needed)
      v8bf p8 = *(const v8bf*)&pls[wave][ln][half * 8];
      v16bf pa;
#pragma unroll
      for (int e = 0; e < 8; ++e) { pa[e] = p8[e]; pa[e + 8] = (bf16)0.0f; }
      o[qi] = wmma_bf16(pa, vb, o[qi]);
    }
  }

#pragma unroll
  for (int qi = 0; qi < 2; ++qi) {
#pragma unroll
    for (int r = 0; r < 8; ++r) {
      int lq = (qt0 + qi) * 16 + r + half * 8;
      if (lq < cnt)
        ctx[(size_t)(st + lq) * EDIM + h * 16 + ln] = (bf16)(o[qi][r] / lr[qi][r]);
    }
  }
}

// ---------------- residual add + LayerNorm (one wave per row) ---------------
__global__ void k_addln(const float* __restrict__ x, const float* __restrict__ a,
                        const float* __restrict__ g, const float* __restrict__ bb,
                        float* __restrict__ yf, bf16* __restrict__ yb) {
  int row  = (blockIdx.x * blockDim.x + threadIdx.x) >> 5;
  int lane = threadIdx.x & 31;
  if (row >= NTOT) return;
  const float* xr = x + (size_t)row * EDIM + lane * 4;
  const float* ar = a + (size_t)row * EDIM + lane * 4;
  float v[4], s = 0.0f, s2 = 0.0f;
#pragma unroll
  for (int j = 0; j < 4; ++j) {
    v[j] = xr[j] + ar[j]; s += v[j]; s2 += v[j] * v[j];
  }
#pragma unroll
  for (int off = 16; off > 0; off >>= 1) {
    s  += __shfl_xor(s,  off, 32);
    s2 += __shfl_xor(s2, off, 32);
  }
  float mean = s * (1.0f / EDIM);
  float var  = s2 * (1.0f / EDIM) - mean * mean;
  float inv  = rsqrtf(var + 1e-5f);
#pragma unroll
  for (int j = 0; j < 4; ++j) {
    int c = lane * 4 + j;
    float ov = (v[j] - mean) * inv * g[c] + bb[c];
    yf[(size_t)row * EDIM + c] = ov;
    if (yb) yb[(size_t)row * EDIM + c] = (bf16)ov;
  }
}

// ---------------- host launcher ---------------------------------------------
static size_t algn(size_t x) { return (x + 255) & ~(size_t)255; }

extern "C" void kernel_launch(void* const* d_in, const int* in_sizes, int n_in,
                              void* d_out, int out_size, void* d_ws, size_t ws_size,
                              hipStream_t stream) {
  (void)in_sizes; (void)n_in; (void)out_size; (void)ws_size;
  const float* pf    = (const float*)d_in[0];
  const int*   vc    = (const int*)  d_in[1];
  const float* pre_w = (const float*)d_in[4];
  const float* pre_b = (const float*)d_in[5];
  const float* in_w  = (const float*)d_in[6];
  const float* in_b  = (const float*)d_in[7];
  const float* out_w = (const float*)d_in[8];
  const float* out_b = (const float*)d_in[9];
  const float* ln1_g = (const float*)d_in[10];
  const float* ln1_b = (const float*)d_in[11];
  const float* w1    = (const float*)d_in[12];
  const float* b1    = (const float*)d_in[13];
  const float* w2    = (const float*)d_in[14];
  const float* b2    = (const float*)d_in[15];
  const float* ln2_g = (const float*)d_in[16];
  const float* ln2_b = (const float*)d_in[17];
  float* out = (float*)d_out;

  // workspace carve-up
  uint8_t* w = (uint8_t*)d_ws;
  size_t off = 0;
  int*  seg    = (int*) (w + off); off = algn(off + 8 * sizeof(int));
  bf16* pf_b   = (bf16*)(w + off); off = algn(off + (size_t)NTOT * CINC * 2);
  bf16* wpre_t = (bf16*)(w + off); off = algn(off + (size_t)EDIM * CINC * 2);
  bf16* win_t  = (bf16*)(w + off); off = algn(off + (size_t)E3 * EDIM * 2);
  bf16* wout_t = (bf16*)(w + off); off = algn(off + (size_t)EDIM * EDIM * 2);
  bf16* w1_t   = (bf16*)(w + off); off = algn(off + (size_t)FF * EDIM * 2);
  bf16* w2_t   = (bf16*)(w + off); off = algn(off + (size_t)EDIM * FF * 2);
  float* x_f   = (float*)(w + off); off = algn(off + (size_t)NTOT * EDIM * 4);
  bf16* x_b    = (bf16*)(w + off); off = algn(off + (size_t)NTOT * EDIM * 2);
  bf16* qkv_b  = (bf16*)(w + off); off = algn(off + (size_t)NTOT * E3 * 2);
  bf16* vt_b   = (bf16*)(w + off); off = algn(off + (size_t)EDIM * BATCH * SMAX * 2);
  bf16* ctx_b  = (bf16*)(w + off); off = algn(off + (size_t)NTOT * EDIM * 2);
  float* att_f = (float*)(w + off); off = algn(off + (size_t)NTOT * EDIM * 4);
  float* h1_f  = (float*)(w + off); off = algn(off + (size_t)NTOT * EDIM * 4);
  bf16* h1_b   = (bf16*)(w + off); off = algn(off + (size_t)NTOT * EDIM * 2);
  bf16* gact_b = (bf16*)(w + off); off = algn(off + (size_t)NTOT * FF * 2);
  float* f2_f  = (float*)(w + off); off = algn(off + (size_t)NTOT * EDIM * 4);

  // 1) convert features/weights to bf16 (weights transposed to [Nout][K])
  k_prep<<<1024, 256, 0, stream>>>(pf, pre_w, in_w, out_w, w1, w2,
                                   pf_b, wpre_t, win_t, wout_t, w1_t, w2_t);
  // 2) per-batch segment bounds
  k_seg<<<1, 256, 0, stream>>>(vc, seg);
  // 3) x = pf @ pre_w + pre_b            [8192,64]x[64,128]
  k_gemm<<<(NTOT/32)*(EDIM/64)/8, 256, 0, stream>>>(
      pf_b, wpre_t, pre_b, x_f, x_b, NTOT, EDIM, CINC, 0);
  // 4) qkv = x @ in_w + in_b             [8192,128]x[128,384]
  k_gemm<<<(NTOT/32)*(E3/64)/8, 256, 0, stream>>>(
      x_b, win_t, in_b, nullptr, qkv_b, NTOT, E3, EDIM, 0);
  // 5) V transpose per head (segment-aligned to SMAX)
  k_vt<<<(NTOT*EDIM)/256, 256, 0, stream>>>(qkv_b, vc, seg, vt_b);
  // 6) flash attention -> ctx (2 q-tiles per wave, shared K/V fragments)
  k_attn<<<BATCH*NHEAD*QG2, 256, 0, stream>>>(qkv_b, vt_b, seg, ctx_b);
  // 7) attn_out = ctx @ out_w + out_b
  k_gemm<<<(NTOT/32)*(EDIM/64)/8, 256, 0, stream>>>(
      ctx_b, wout_t, out_b, att_f, nullptr, NTOT, EDIM, EDIM, 0);
  // 8) h1 = LN(x + attn_out)
  k_addln<<<NTOT/8, 256, 0, stream>>>(x_f, att_f, ln1_g, ln1_b, h1_f, h1_b);
  // 9) g = gelu(h1 @ w1 + b1)
  k_gemm<<<(NTOT/32)*(FF/64)/8, 256, 0, stream>>>(
      h1_b, w1_t, b1, nullptr, gact_b, NTOT, FF, EDIM, 1);
  // 10) ffn = g @ w2 + b2
  k_gemm<<<(NTOT/32)*(EDIM/64)/8, 256, 0, stream>>>(
      gact_b, w2_t, b2, f2_f, nullptr, NTOT, EDIM, FF, 0);
  // 11) out = LN(h1 + ffn)   (compact rows == ragged output order)
  k_addln<<<NTOT/8, 256, 0, stream>>>(h1_f, f2_f, ln2_g, ln2_b, out, nullptr);
}
